// CausalSelfAttention_10995116278204
// MI455X (gfx1250) — matrix operand order repaired and checked by reading, not compiled, not run on hardware
//
#include <hip/hip_runtime.h>

typedef __bf16 bf16_t;
typedef __attribute__((ext_vector_type(16))) __bf16 v16bf;
typedef __attribute__((ext_vector_type(8)))  __bf16 v8bf;
typedef __attribute__((ext_vector_type(8)))  float  v8f;
typedef __attribute__((ext_vector_type(4)))  int    v4i;

#define BB 4
#define TT 2048
#define CC 1024
#define HH 16
#define DD 64

#ifndef __has_builtin
#define __has_builtin(x) 0
#endif

#if __has_builtin(__builtin_amdgcn_global_load_async_to_lds_b128)
#define HAVE_ASYNC_LDS 1
#else
#define HAVE_ASYNC_LDS 0
#endif

typedef __attribute__((address_space(1))) v4i* gv4p;   // global int4*
typedef __attribute__((address_space(3))) v4i* lv4p;   // LDS int4*

__device__ __forceinline__ void async_copy16(const void* g, void* l) {
#if HAVE_ASYNC_LDS
  __builtin_amdgcn_global_load_async_to_lds_b128((gv4p)(void*)g, (lv4p)l, 0, 0);
#endif
}

__device__ __forceinline__ void async_wait0() {
#if HAVE_ASYNC_LDS
#if __has_builtin(__builtin_amdgcn_s_wait_asynccnt)
  __builtin_amdgcn_s_wait_asynccnt(0);
#else
  asm volatile("s_wait_asynccnt 0x0" ::: "memory");
#endif
#endif
}

// ---------------------------------------------------------------- helpers ---

__device__ __forceinline__ v8f wmma_bf16(v16bf a, v16bf b, v8f c) {
  // D = A(16x32 bf16) * B(32x16 bf16) + C(f32)
  return __builtin_amdgcn_wmma_f32_16x16x32_bf16(
      /*neg_a=*/false, a, /*neg_b=*/false, b,
      /*c_mod=*/(short)0, c, /*reuse_a=*/false, /*reuse_b=*/false);
}

// A-fragment: lane holds row M = lane%16; elements e<8 -> K=kbase+e,
// e>=8 -> K=kbase+16+(e-8), kbase = 32*kk + 8*(lane>=16).
__device__ __forceinline__ v16bf load_a_frag(const bf16_t* rowptr, int kbase) {
  v8bf lo = *(const v8bf*)(rowptr + kbase);
  v8bf hi = *(const v8bf*)(rowptr + kbase + 16);
  v16bf a;
#pragma unroll
  for (int i = 0; i < 8; ++i) { a[i] = lo[i]; a[i + 8] = hi[i]; }
  return a;
}

// B-fragment: lane holds col N = lane%16; elements e<8 -> K=kbase+e,
// e>=8 -> K=kbase+16+(e-8), kbase = 32*kk + 8*(lane>=16).
__device__ __forceinline__ v16bf load_b_frag(const bf16_t* p) {
  v8bf lo = *(const v8bf*)p;
  v8bf hi = *(const v8bf*)(p + 16);
  v16bf b;
#pragma unroll
  for (int i = 0; i < 8; ++i) { b[i] = lo[i]; b[i + 8] = hi[i]; }
  return b;
}

__device__ __forceinline__ float halfmax(float v) {
#pragma unroll
  for (int m = 1; m < 16; m <<= 1) v = fmaxf(v, __shfl_xor(v, m, 32));
  return v;
}
__device__ __forceinline__ float halfsum(float v) {
#pragma unroll
  for (int m = 1; m < 16; m <<= 1) v += __shfl_xor(v, m, 32);
  return v;
}

// ---------------------------------------------------------------- convert ---

__global__ void cvt_f32_to_bf16(const float* __restrict__ in,
                                bf16_t* __restrict__ out, int n) {
  int i = blockIdx.x * blockDim.x + threadIdx.x;
  int stride = blockDim.x * gridDim.x;
  for (; i < n; i += stride) out[i] = (bf16_t)in[i];
}

// ------------------------------------------------------------------- GEMM ---
// C[M,N] = A[M,K](bf16,rm) * W[K,N](bf16,rm) + bias(f32)
// MODE 0: out = fp32 row-major [M,N]
// MODE 1: out = bf16 in [B,H,T,D] layout (row -> (b,t), col -> (h,d))
// Block: 128 threads = 4 waves. Block tile 128(M) x 64(N), K-tile 64.

template <int MODE>
__global__ void __launch_bounds__(128)
gemm_wmma(const bf16_t* __restrict__ A, const bf16_t* __restrict__ W,
          const float* __restrict__ bias, void* __restrict__ out,
          int M, int N, int K) {
  __shared__ bf16_t Alds[128][72];   // [m][k], padded
  __shared__ bf16_t Wt[64][72];      // [n][k] transposed, padded

  const int lane   = threadIdx.x & 31;
  const int wave   = threadIdx.x >> 5;
  const int laneHi = lane >> 4;
  const int lane16 = lane & 15;
  const int n0 = blockIdx.x * 64;
  const int m0 = blockIdx.y * 128;

  v8f acc[2][4];
#pragma unroll
  for (int mi = 0; mi < 2; ++mi)
#pragma unroll
    for (int ni = 0; ni < 4; ++ni)
#pragma unroll
      for (int r = 0; r < 8; ++r) acc[mi][ni][r] = 0.0f;

  for (int k0 = 0; k0 < K; k0 += 64) {
    // ---- stage A tile [128][64] ----
#if HAVE_ASYNC_LDS
    {
      // 1024 chunks of 16B (8 bf16); 128 threads x 8 chunks, lane-contiguous
#pragma unroll
      for (int i = 0; i < 8; ++i) {
        const int chunk = i * 128 + threadIdx.x;
        const int row = chunk >> 3;
        const int col = (chunk & 7) * 8;
        async_copy16(A + (size_t)(m0 + row) * K + k0 + col, &Alds[row][col]);
      }
    }
#else
    {
      int row = threadIdx.x;
      const bf16_t* src = A + (size_t)(m0 + row) * K + k0;
#pragma unroll
      for (int s = 0; s < 4; ++s)
        *(v16bf*)&Alds[row][s * 16] = *(const v16bf*)(src + s * 16);
    }
#endif
    // ---- stage W tile transposed -> Wt[n][k] ----
    {
      int k = threadIdx.x >> 1;
      int seg = (threadIdx.x & 1) * 32;
      const bf16_t* src = W + (size_t)(k0 + k) * N + n0 + seg;
      if (k0 + 64 < K) {
        __builtin_prefetch(src + (size_t)64 * N, 0, 1);       // next W k-rows
        __builtin_prefetch(A + (size_t)(m0 + threadIdx.x) * K + k0 + 64, 0, 1);
      }
      v16bf w0 = *(const v16bf*)src;
      v16bf w1 = *(const v16bf*)(src + 16);
#pragma unroll
      for (int i = 0; i < 16; ++i) Wt[seg + i][k] = w0[i];
#pragma unroll
      for (int i = 0; i < 16; ++i) Wt[seg + 16 + i][k] = w1[i];
    }
    async_wait0();
    __syncthreads();

#pragma unroll
    for (int kk = 0; kk < 2; ++kk) {
      v16bf a[2];
#pragma unroll
      for (int mi = 0; mi < 2; ++mi)
        a[mi] = load_a_frag(&Alds[wave * 32 + mi * 16 + lane16][0],
                            kk * 32 + laneHi * 8);
#pragma unroll
      for (int ni = 0; ni < 4; ++ni) {
        v16bf b = load_b_frag(&Wt[ni * 16 + lane16][kk * 32 + laneHi * 8]);
#pragma unroll
        for (int mi = 0; mi < 2; ++mi)
          acc[mi][ni] = wmma_bf16(a[mi], b, acc[mi][ni]);
      }
    }
    __syncthreads();
  }

  // epilogue: +bias, store
#pragma unroll
  for (int ni = 0; ni < 4; ++ni) {
    const int col = n0 + ni * 16 + lane16;
    const float bv = bias[col];
#pragma unroll
    for (int mi = 0; mi < 2; ++mi) {
#pragma unroll
      for (int r = 0; r < 8; ++r) {
        const int row = m0 + wave * 32 + mi * 16 + r + laneHi * 8;
        const float v = acc[mi][ni][r] + bv;
        if (MODE == 0) {
          ((float*)out)[(size_t)row * N + col] = v;
        } else {
          const int b = row / TT, t = row % TT;
          const int h = col >> 6, d = col & 63;
          ((bf16_t*)out)[(((size_t)b * HH + h) * TT + t) * DD + d] = (bf16_t)v;
        }
      }
    }
  }
}

// -------------------------------------------------------- flash attention ---
// One block per (b, h, 128 q-rows). 4 waves; wave w owns q-rows [q0b+32w, +32).

__global__ void __launch_bounds__(128)
attn_kernel(const bf16_t* __restrict__ Q, const bf16_t* __restrict__ Kg,
            const bf16_t* __restrict__ V, bf16_t* __restrict__ Y) {
  __shared__ bf16_t Klds[64][72];       // [key][d]
  __shared__ bf16_t Vt[64][72];         // [d][key] (transposed)
  __shared__ bf16_t Plds[4][32][72];    // per-wave P scratch [q][key]

  const int lane   = threadIdx.x & 31;
  const int wave   = threadIdx.x >> 5;
  const int laneHi = lane >> 4;
  const int lane16 = lane & 15;

  const int nqc = TT / 128;
  const int bh  = blockIdx.x / nqc;
  const int qc  = blockIdx.x % nqc;
  const int q0b = qc * 128;
  const int q0w = q0b + wave * 32;
  const size_t base = (size_t)bh * TT * DD;

  const float sc = 0.125f * 1.44269504f;  // 1/sqrt(64) * log2(e)

  // Q fragments (resident for the whole kernel)
  v16bf qa[2][2];
#pragma unroll
  for (int mi = 0; mi < 2; ++mi)
#pragma unroll
    for (int kk = 0; kk < 2; ++kk)
      qa[mi][kk] = load_a_frag(Q + base + (size_t)(q0w + mi * 16 + lane16) * DD,
                               kk * 32 + laneHi * 8);

  v8f o[2][4];
  float mst[2][8], lst[2][8];
#pragma unroll
  for (int mi = 0; mi < 2; ++mi) {
#pragma unroll
    for (int r = 0; r < 8; ++r) { mst[mi][r] = -1.0e30f; lst[mi][r] = 0.0f; }
#pragma unroll
    for (int ni = 0; ni < 4; ++ni)
#pragma unroll
      for (int r = 0; r < 8; ++r) o[mi][ni][r] = 0.0f;
  }

  const int jend = q0b + 128;           // causal: keys up to block's last q row
  for (int j = 0; j < jend; j += 64) {
    // ---- cooperative stage of K (row-major, async) and V (transposed) ----
#if HAVE_ASYNC_LDS
    {
      // K tile: 512 chunks of 16B; 128 threads x 4 chunks
#pragma unroll
      for (int i = 0; i < 4; ++i) {
        const int chunk = i * 128 + threadIdx.x;
        const int key = chunk >> 3;
        const int col = (chunk & 7) * 8;
        async_copy16(Kg + base + (size_t)(j + key) * DD + col, &Klds[key][col]);
      }
    }
#else
    {
      const int key = threadIdx.x >> 1;
      const int seg = (threadIdx.x & 1) * 32;
      const bf16_t* ks = Kg + base + (size_t)(j + key) * DD + seg;
      *(v16bf*)&Klds[key][seg]      = *(const v16bf*)ks;
      *(v16bf*)&Klds[key][seg + 16] = *(const v16bf*)(ks + 16);
    }
#endif
    {
      const int key = threadIdx.x >> 1;
      const int seg = (threadIdx.x & 1) * 32;
      const bf16_t* vs = V + base + (size_t)(j + key) * DD + seg;
      if (j + 64 < jend)
        __builtin_prefetch(vs + (size_t)64 * DD, 0, 1);       // next V tile
      v16bf v0 = *(const v16bf*)vs;
      v16bf v1 = *(const v16bf*)(vs + 16);
#pragma unroll
      for (int i = 0; i < 16; ++i) Vt[seg + i][key]      = v0[i];
#pragma unroll
      for (int i = 0; i < 16; ++i) Vt[seg + 16 + i][key] = v1[i];
    }
    async_wait0();
    __syncthreads();

    if (j < q0w + 32) {   // wave-uniform: this wave still has work at this j
      // ---- S = Q * K^T ----
      v8f s[2][4];
#pragma unroll
      for (int mi = 0; mi < 2; ++mi)
#pragma unroll
        for (int ni = 0; ni < 4; ++ni)
#pragma unroll
          for (int r = 0; r < 8; ++r) s[mi][ni][r] = 0.0f;

#pragma unroll
      for (int kk = 0; kk < 2; ++kk) {
#pragma unroll
        for (int ni = 0; ni < 4; ++ni) {
          v16bf kb = load_b_frag(&Klds[ni * 16 + lane16][kk * 32 + laneHi * 8]);
#pragma unroll
          for (int mi = 0; mi < 2; ++mi)
            s[mi][ni] = wmma_bf16(qa[mi][kk], kb, s[mi][ni]);
        }
      }

      // ---- scale + causal mask ----
#pragma unroll
      for (int mi = 0; mi < 2; ++mi)
#pragma unroll
        for (int ni = 0; ni < 4; ++ni)
#pragma unroll
          for (int r = 0; r < 8; ++r) {
            const int qrow = q0w + mi * 16 + r + laneHi * 8;
            const int key  = j + ni * 16 + lane16;
            const float v = s[mi][ni][r] * sc;
            s[mi][ni][r] = (key > qrow) ? -1.0e30f : v;
          }

      // ---- online softmax (base-2) ----
      float alpha[2][8];
#pragma unroll
      for (int mi = 0; mi < 2; ++mi)
#pragma unroll
        for (int r = 0; r < 8; ++r) {
          float mx = s[mi][0][r];
          mx = fmaxf(mx, s[mi][1][r]);
          mx = fmaxf(mx, s[mi][2][r]);
          mx = fmaxf(mx, s[mi][3][r]);
          mx = halfmax(mx);
          const float mn = fmaxf(mst[mi][r], mx);
          alpha[mi][r] = exp2f(mst[mi][r] - mn);
          mst[mi][r] = mn;
        }

#pragma unroll
      for (int mi = 0; mi < 2; ++mi)
#pragma unroll
        for (int r = 0; r < 8; ++r) {
          float rsum = 0.0f;
#pragma unroll
          for (int ni = 0; ni < 4; ++ni) {
            const float p = exp2f(s[mi][ni][r] - mst[mi][r]);
            s[mi][ni][r] = p;
            rsum += p;
          }
          rsum = halfsum(rsum);
          lst[mi][r] = lst[mi][r] * alpha[mi][r] + rsum;
        }

#pragma unroll
      for (int mi = 0; mi < 2; ++mi)
#pragma unroll
        for (int ni = 0; ni < 4; ++ni)
#pragma unroll
          for (int r = 0; r < 8; ++r) o[mi][ni][r] *= alpha[mi][r];

      // ---- P -> per-wave LDS (D-layout -> A-layout re-tiling) ----
#pragma unroll
      for (int mi = 0; mi < 2; ++mi)
#pragma unroll
        for (int ni = 0; ni < 4; ++ni)
#pragma unroll
          for (int r = 0; r < 8; ++r)
            Plds[wave][mi * 16 + r + laneHi * 8][ni * 16 + lane16] =
                (bf16_t)s[mi][ni][r];

      // ---- O += P * V ----
#pragma unroll
      for (int kk = 0; kk < 2; ++kk) {
        v16bf pa[2];
#pragma unroll
        for (int mi = 0; mi < 2; ++mi)
          pa[mi] = load_a_frag(&Plds[wave][mi * 16 + lane16][0],
                               kk * 32 + laneHi * 8);
#pragma unroll
        for (int ni = 0; ni < 4; ++ni) {
          v16bf vb = load_b_frag(&Vt[ni * 16 + lane16][kk * 32 + laneHi * 8]);
#pragma unroll
          for (int mi = 0; mi < 2; ++mi)
            o[mi][ni] = wmma_bf16(pa[mi], vb, o[mi][ni]);
        }
      }
    }
    __syncthreads();
  }

  // ---- epilogue: O / l -> Y bf16 [B*T, C] ----
  const int b = bh / HH, h = bh % HH;
#pragma unroll
  for (int mi = 0; mi < 2; ++mi)
#pragma unroll
    for (int r = 0; r < 8; ++r) {
      const float inv = 1.0f / lst[mi][r];
      const int qrow = q0w + mi * 16 + r + laneHi * 8;
#pragma unroll
      for (int ni = 0; ni < 4; ++ni) {
        const int d = ni * 16 + lane16;
        Y[((size_t)b * TT + qrow) * CC + h * DD + d] =
            (bf16_t)(o[mi][ni][r] * inv);
      }
    }
}

// ----------------------------------------------------------------- launch ---

extern "C" void kernel_launch(void* const* d_in, const int* in_sizes, int n_in,
                              void* d_out, int out_size, void* d_ws,
                              size_t ws_size, hipStream_t stream) {
  (void)in_sizes; (void)n_in; (void)out_size; (void)ws_size;

  const float* x  = (const float*)d_in[0];
  const float* Wq = (const float*)d_in[1];
  const float* bq = (const float*)d_in[2];
  const float* Wk = (const float*)d_in[3];
  const float* bk = (const float*)d_in[4];
  const float* Wv = (const float*)d_in[5];
  const float* bv = (const float*)d_in[6];
  const float* Wo = (const float*)d_in[7];
  const float* bo = (const float*)d_in[8];

  char* ws = (char*)d_ws;
  size_t off = 0;
  auto carve = [&](size_t bytes) -> char* {
    char* p = ws + off;
    off += (bytes + 255) & ~(size_t)255;
    return p;
  };
  const size_t nx = (size_t)BB * TT * CC;  // 8M activations
  const size_t nw = (size_t)CC * CC;       // 1M weights
  bf16_t* xb  = (bf16_t*)carve(nx * 2);
  bf16_t* wqb = (bf16_t*)carve(nw * 2);
  bf16_t* wkb = (bf16_t*)carve(nw * 2);
  bf16_t* wvb = (bf16_t*)carve(nw * 2);
  bf16_t* wob = (bf16_t*)carve(nw * 2);
  bf16_t* Qb  = (bf16_t*)carve(nx * 2);
  bf16_t* Kb  = (bf16_t*)carve(nx * 2);
  bf16_t* Vb  = (bf16_t*)carve(nx * 2);
  bf16_t* Yb  = (bf16_t*)carve(nx * 2);

  cvt_f32_to_bf16<<<1024, 256, 0, stream>>>(x, xb, (int)nx);
  cvt_f32_to_bf16<<<256, 256, 0, stream>>>(Wq, wqb, (int)nw);
  cvt_f32_to_bf16<<<256, 256, 0, stream>>>(Wk, wkb, (int)nw);
  cvt_f32_to_bf16<<<256, 256, 0, stream>>>(Wv, wvb, (int)nw);
  cvt_f32_to_bf16<<<256, 256, 0, stream>>>(Wo, wob, (int)nw);

  dim3 ggrid(CC / 64, (BB * TT) / 128);
  gemm_wmma<1><<<ggrid, 128, 0, stream>>>(xb, wqb, bq, Qb, BB * TT, CC, CC);
  gemm_wmma<1><<<ggrid, 128, 0, stream>>>(xb, wkb, bk, Kb, BB * TT, CC, CC);
  gemm_wmma<1><<<ggrid, 128, 0, stream>>>(xb, wvb, bv, Vb, BB * TT, CC, CC);

  attn_kernel<<<BB * HH * (TT / 128), 128, 0, stream>>>(Qb, Kb, Vb, Yb);

  gemm_wmma<0><<<ggrid, 128, 0, stream>>>(Yb, wob, bo, d_out, BB * TT, CC, CC);
}
